// RGIN_25786983645586
// MI455X (gfx1250) — matrix operand order repaired and checked by reading, compiled, not run on hardware
//
#include <hip/hip_runtime.h>
#include <stdint.h>
#include <stddef.h>

// Problem constants (match reference file)
#define NN 50000
#define EE 640000
#define DD 128
#define OO 128
#define RR 8
#define ROW_TILES (NN / 16)   // 3125

typedef __attribute__((ext_vector_type(16))) __bf16   v16bf;
typedef __attribute__((ext_vector_type(8)))  float    v8f;
typedef __attribute__((ext_vector_type(4)))  uint32_t u32x4;

// TDM descriptor vector types (probe-verified builtin signatures)
typedef __attribute__((ext_vector_type(4))) uint32_t tdm_u32x4;
typedef __attribute__((ext_vector_type(8))) int      tdm_i32x8;
typedef __attribute__((ext_vector_type(4))) int      tdm_i32x4;

union Frag16 {
    v16bf    v;
    u32x4    q[2];
    uint32_t u[8];
};

__device__ __forceinline__ uint16_t f32_to_bf16(float f) {
    uint32_t u = __builtin_bit_cast(uint32_t, f);
    u += 0x7fffu + ((u >> 16) & 1u);   // round-to-nearest-even
    return (uint16_t)(u >> 16);
}
__device__ __forceinline__ float bf16_to_f32(uint32_t lo16) {
    return __builtin_bit_cast(float, lo16 << 16);
}

// Generic LDS pointer -> byte offset within the workgroup LDS allocation
// (AS3->generic addrspacecast puts the DS offset in the low 32 bits).
__device__ __forceinline__ uint32_t lds_byte_off(const void* p) {
    return (uint32_t)(uintptr_t)p;
}

// Tensor Data Mover: 1-D DMA of `ndwords` dwords global -> LDS.
// D# per ISA cdna5 §8.3/8.4: group0 = {count=1 | lds_addr | global_addr | type=2},
// group1 = {data_size=4B, tensor_dim0=tile_dim0=ndwords, tensor_dim1=tile_dim1=1,
//           tensor_dim0_stride=ndwords}.  Groups 2/3 zero (<=2-D tensor).
__device__ __forceinline__ void tdm_load_to_lds(uint32_t lds_off, const void* gptr,
                                                uint32_t ndwords) {
    uint64_t ga = (uint64_t)(uintptr_t)gptr;
    tdm_u32x4 g0 = { 1u,                                   // count=1 (valid D#)
                     lds_off,                              // [63:32] lds_addr
                     (uint32_t)ga,                         // [95:64] global_addr lo
                     ((uint32_t)(ga >> 32) & 0x01ffffffu)  // [120:96] global_addr hi
                         | (2u << 30) };                   // [127:126] type=2
    tdm_i32x8 g1 = { (int)0x00020000,                      // data_size=2 (4 bytes)
                     (int)(ndwords << 16),                 // tensor_dim0[15:0]
                     (int)0x00010000,                      // tensor_dim1=1
                     (int)(ndwords << 16),                 // tile_dim0
                     1,                                    // tile_dim1=1
                     (int)ndwords,                         // tensor_dim0_stride lo
                     0, 0 };
    tdm_i32x4 z4 = {0, 0, 0, 0};
#if __clang_major__ >= 23
    tdm_i32x8 z8 = {0, 0, 0, 0, 0, 0, 0, 0};
    __builtin_amdgcn_tensor_load_to_lds(g0, g1, z4, z4, z8, 0);
#else
    __builtin_amdgcn_tensor_load_to_lds(g0, g1, z4, z4, 0);
#endif
}

// A fragment (16x32 bf16, ISA 7.12.2): lane = half*16 + row; half 0 holds
// K = k0..k0+7 and k0+16..k0+23, half 1 holds K = k0+8..15 and k0+24..31.
// Two 16B loads; works for global (xb) and LDS (h tile) pointers.
__device__ __forceinline__ Frag16 load_a(const uint16_t* base, int ld, int k0, int lane) {
    const int row = lane & 15, half = lane >> 4;
    const uint16_t* p = base + row * ld + k0 + half * 8;
    Frag16 a;
    a.q[0] = *(const u32x4*)(p);
    a.q[1] = *(const u32x4*)(p + 16);
    return a;
}

// B fragment from pre-swizzled weights (global or LDS): each lane reads its
// 8 fragment dwords as 32 contiguous bytes -> two (global|ds)_load_b128.
__device__ __forceinline__ Frag16 load_b(const uint32_t* mat, int nt, int s, int lane) {
    const uint32_t* p = mat + ((((nt << 2) + s) << 5) + lane) * 8;
    Frag16 b;
    b.q[0] = *(const u32x4*)(p);
    b.q[1] = *(const u32x4*)(p + 4);
    return b;
}

__device__ __forceinline__ v8f wmma_bf16(const Frag16& a, const Frag16& b, v8f c) {
    return __builtin_amdgcn_wmma_f32_16x16x32_bf16(false, a.v, false, b.v,
                                                   (short)0, c, false, false);
}

// ---------------- conversion / packing ----------------

__global__ void __launch_bounds__(256)
k_cvt_x(const float* __restrict__ x, uint32_t* __restrict__ xb) {
    int t = blockIdx.x * 256 + threadIdx.x;           // one dword = 2 elements
    if (t >= NN * DD / 2) return;
    float a = x[2 * t], b = x[2 * t + 1];
    xb[t] = (uint32_t)f32_to_bf16(a) | ((uint32_t)f32_to_bf16(b) << 16);
}

// Pack a [128][128] f32 row-major (K,N) matrix into B-fragment order:
// dst dword = ((nt*4+s)*32 + lane)*8 + j ; K = s*32 + half*8 + (j&3)*2 + (j>>2)*16 (+1),
// N = nt*16 + (lane&15). 11 matrices: 8 relations + w1_W + m1_W + m2_W.
__global__ void __launch_bounds__(256)
k_pack(const float* __restrict__ wrel, const float* __restrict__ w1,
       const float* __restrict__ m1,   const float* __restrict__ m2,
       uint32_t* __restrict__ o_wrel, uint32_t* __restrict__ o_w1,
       uint32_t* __restrict__ o_m1,   uint32_t* __restrict__ o_m2) {
    int t = blockIdx.x * 256 + threadIdx.x;           // 11 * 8192 threads
    int mat = t >> 13;
    int idx = t & 8191;
    const float* src; uint32_t* dp;
    if (mat < 8)       { src = wrel + mat * (DD * OO); dp = o_wrel + mat * 8192; }
    else if (mat == 8) { src = w1; dp = o_w1; }
    else if (mat == 9) { src = m1; dp = o_m1; }
    else               { src = m2; dp = o_m2; }
    int j    = idx & 7;
    int lane = (idx >> 3) & 31;
    int s    = (idx >> 8) & 3;
    int nt   = idx >> 10;
    int col  = nt * 16 + (lane & 15);
    int half = lane >> 4;
    int k0   = s * 32 + half * 8 + (j & 3) * 2 + (j >> 2) * 16;
    uint32_t lo = f32_to_bf16(src[k0 * OO + col]);
    uint32_t hi = f32_to_bf16(src[(k0 + 1) * OO + col]);
    dp[idx] = lo | (hi << 16);
}

// ---------------- pass 1: h_all = x @ W_rel  (bf16 out) ----------------
// grid (391, 16): blockIdx.y = 64-column group (cg); TDM pulls that group's
// 16KB packed weight slab into LDS once per block; 8 waves each do a
// 16-row x 64-col tile with A reused across 4 column tiles.
__global__ void __launch_bounds__(256)
k_project(const uint16_t* __restrict__ xb, const uint32_t* __restrict__ wrelb,
          uint16_t* __restrict__ hall) {
    __shared__ uint32_t wslab[4096];                  // 16 KB
    const int cg = blockIdx.y;                        // 0..15
    if (threadIdx.x == 0) {
        const uint32_t* gsrc = wrelb + (cg >> 1) * 8192 + (cg & 1) * 4096;
        tdm_load_to_lds(lds_byte_off(wslab), gsrc, 4096);
    }
    __builtin_amdgcn_s_wait_tensorcnt(0);
    __syncthreads();

    const int w = threadIdx.x >> 5, lane = threadIdx.x & 31;
    const int row_tile = blockIdx.x * 8 + w;
    if (row_tile >= ROW_TILES) return;                // wave-uniform, after barrier
    const int m0 = row_tile * 16;

    const v8f z = {0.f, 0.f, 0.f, 0.f, 0.f, 0.f, 0.f, 0.f};
    v8f c[4] = {z, z, z, z};
    const uint16_t* xrow = xb + (size_t)m0 * DD;
#pragma unroll
    for (int s = 0; s < 4; ++s) {
        Frag16 a = load_a(xrow, DD, s * 32, lane);
#pragma unroll
        for (int t = 0; t < 4; ++t) {
            Frag16 b = load_b(wslab, t, s, lane);     // ds_load_b128 x2
            c[t] = wmma_bf16(a, b, c[t]);
        }
    }
    const int col = lane & 15, half = lane >> 4;
#pragma unroll
    for (int t = 0; t < 4; ++t)
#pragma unroll
        for (int i = 0; i < 8; ++i)
            hall[(size_t)(m0 + i + half * 8) * (RR * OO) + cg * 64 + t * 16 + col] =
                f32_to_bf16(c[t][i]);
}

// ---------------- pass 2: agg[dst] += h_all[src, etype] ----------------
// One wave per edge; lane handles 4 consecutive floats (256B/wave coalesced
// bf16 reads, native f32 global atomics). Bandwidth-critical pass.
__global__ void __launch_bounds__(256)
k_edges(const int* __restrict__ src, const int* __restrict__ dst,
        const int* __restrict__ et,  const uint16_t* __restrict__ hall,
        float* __restrict__ agg) {
    const int g = blockIdx.x * 256 + threadIdx.x;
    const int e = g >> 5;
    const int lane = g & 31;
    if (e >= EE) return;
    const int sn = src[e], dn = dst[e], r = et[e];
    const uint16_t* mp = hall + (size_t)sn * (RR * OO) + r * OO + lane * 4;
    uint32_t p0 = *(const uint32_t*)(mp);
    uint32_t p1 = *(const uint32_t*)(mp + 2);
    float f0 = bf16_to_f32(p0 & 0xffffu), f1 = bf16_to_f32(p0 >> 16);
    float f2 = bf16_to_f32(p1 & 0xffffu), f3 = bf16_to_f32(p1 >> 16);
    float* ap = agg + (size_t)dn * OO + lane * 4;
    __hip_atomic_fetch_add(ap + 0, f0, __ATOMIC_RELAXED, __HIP_MEMORY_SCOPE_AGENT);
    __hip_atomic_fetch_add(ap + 1, f1, __ATOMIC_RELAXED, __HIP_MEMORY_SCOPE_AGENT);
    __hip_atomic_fetch_add(ap + 2, f2, __ATOMIC_RELAXED, __HIP_MEMORY_SCOPE_AGENT);
    __hip_atomic_fetch_add(ap + 3, f3, __ATOMIC_RELAXED, __HIP_MEMORY_SCOPE_AGENT);
}

// ---------------- pass 3: fused  out = relu((x@w1+b1+agg)@m1+bm1)@m2+bm2 ----
// 4 waves/block, 16 rows/wave. One 32KB LDS weight slab is TDM-reloaded per
// stage (w1 -> m1 -> m2); 16x128 intermediates round-trip through a
// wave-private LDS slice for the C->A re-layout (DS in-order per wave).
__global__ void __launch_bounds__(128)
k_mlp(const uint16_t* __restrict__ xb,
      const uint32_t* __restrict__ w1b, const uint32_t* __restrict__ m1b,
      const uint32_t* __restrict__ m2b,
      const float* __restrict__ b1, const float* __restrict__ bm1,
      const float* __restrict__ bm2,
      const float* __restrict__ agg, float* __restrict__ out) {
    __shared__ uint32_t wslab[8192];          // 32 KB: current stage's weights
    __shared__ uint16_t hbuf[4][16 * DD];     // 16 KB: per-wave intermediates
    const int w    = threadIdx.x >> 5;
    const int lane = threadIdx.x & 31;
    const int gw   = blockIdx.x * 4 + w;
    const bool active = (gw < ROW_TILES);     // wave-uniform
    const int m0   = gw * 16;
    const int col  = lane & 15, half = lane >> 4;
    uint16_t* hb = hbuf[w];
    v8f c[8];

    // ---- stage 1: C = agg + b1 ; C += x @ w1 ----
    if (threadIdx.x == 0) tdm_load_to_lds(lds_byte_off(wslab), w1b, 8192);
    __builtin_amdgcn_s_wait_tensorcnt(0);
    __syncthreads();
    if (active) {
#pragma unroll
        for (int j = 0; j < 8; ++j) {
            float bj = b1[j * 16 + col];
#pragma unroll
            for (int i = 0; i < 8; ++i)
                c[j][i] = agg[(size_t)(m0 + i + half * 8) * OO + j * 16 + col] + bj;
        }
        const uint16_t* xrow = xb + (size_t)m0 * DD;
#pragma unroll
        for (int s = 0; s < 4; ++s) {
            Frag16 a = load_a(xrow, DD, s * 32, lane);
#pragma unroll
            for (int j = 0; j < 8; ++j) {
                Frag16 b = load_b(wslab, j, s, lane);
                c[j] = wmma_bf16(a, b, c[j]);
            }
        }
#pragma unroll
        for (int j = 0; j < 8; ++j)
#pragma unroll
            for (int i = 0; i < 8; ++i)
                hb[(i + half * 8) * DD + j * 16 + col] = f32_to_bf16(c[j][i]);
    }
    __syncthreads();                           // all waves done reading w1 slab

    // ---- stage 2: C = bm1 ; C += h0 @ m1 ; relu ----
    if (threadIdx.x == 0) tdm_load_to_lds(lds_byte_off(wslab), m1b, 8192);
    __builtin_amdgcn_s_wait_tensorcnt(0);
    __syncthreads();
    if (active) {
#pragma unroll
        for (int j = 0; j < 8; ++j) {
            float bj = bm1[j * 16 + col];
#pragma unroll
            for (int i = 0; i < 8; ++i) c[j][i] = bj;
        }
#pragma unroll
        for (int s = 0; s < 4; ++s) {
            Frag16 a = load_a(hb, DD, s * 32, lane);
#pragma unroll
            for (int j = 0; j < 8; ++j) {
                Frag16 b = load_b(wslab, j, s, lane);
                c[j] = wmma_bf16(a, b, c[j]);
            }
        }
#pragma unroll
        for (int j = 0; j < 8; ++j)
#pragma unroll
            for (int i = 0; i < 8; ++i)
                hb[(i + half * 8) * DD + j * 16 + col] =
                    f32_to_bf16(fmaxf(c[j][i], 0.f));
    }
    __syncthreads();

    // ---- stage 3: C = bm2 ; C += h1 @ m2 -> out (f32) ----
    if (threadIdx.x == 0) tdm_load_to_lds(lds_byte_off(wslab), m2b, 8192);
    __builtin_amdgcn_s_wait_tensorcnt(0);
    __syncthreads();
    if (active) {
#pragma unroll
        for (int j = 0; j < 8; ++j) {
            float bj = bm2[j * 16 + col];
#pragma unroll
            for (int i = 0; i < 8; ++i) c[j][i] = bj;
        }
#pragma unroll
        for (int s = 0; s < 4; ++s) {
            Frag16 a = load_a(hb, DD, s * 32, lane);
#pragma unroll
            for (int j = 0; j < 8; ++j) {
                Frag16 b = load_b(wslab, j, s, lane);
                c[j] = wmma_bf16(a, b, c[j]);
            }
        }
#pragma unroll
        for (int j = 0; j < 8; ++j)
#pragma unroll
            for (int i = 0; i < 8; ++i)
                out[(size_t)(m0 + i + half * 8) * OO + j * 16 + col] = c[j][i];
    }
}

extern "C" void kernel_launch(void* const* d_in, const int* in_sizes, int n_in,
                              void* d_out, int out_size, void* d_ws, size_t ws_size,
                              hipStream_t stream) {
    const float* x    = (const float*)d_in[0];
    const int*   src  = (const int*)d_in[1];
    const int*   dst  = (const int*)d_in[2];
    const int*   et   = (const int*)d_in[3];
    const float* wrel = (const float*)d_in[4];
    const float* w1W  = (const float*)d_in[5];
    const float* w1b_ = (const float*)d_in[6];
    const float* m1W  = (const float*)d_in[7];
    const float* m1b_ = (const float*)d_in[8];
    const float* m2W  = (const float*)d_in[9];
    const float* m2b_ = (const float*)d_in[10];
    float* out = (float*)d_out;

    // workspace carve-out (~141 MB total), 256B aligned slabs
    char* ws = (char*)d_ws;
    size_t off = 0;
    auto carve = [&](size_t bytes) {
        char* p = ws + off;
        off = (off + bytes + 255) & ~(size_t)255;
        return p;
    };
    uint16_t* xb    = (uint16_t*)carve((size_t)NN * DD * 2);          // 12.8 MB
    uint32_t* wrelb = (uint32_t*)carve((size_t)RR * 8192 * 4);        // 256 KB
    uint32_t* w1b   = (uint32_t*)carve(8192 * 4);                     // 32 KB
    uint32_t* m1b   = (uint32_t*)carve(8192 * 4);
    uint32_t* m2b   = (uint32_t*)carve(8192 * 4);
    uint16_t* hall  = (uint16_t*)carve((size_t)NN * RR * OO * 2);     // 102.4 MB
    float*    agg   = (float*)carve((size_t)NN * OO * 4);             // 25.6 MB
    (void)ws_size; (void)in_sizes; (void)n_in; (void)out_size;

    hipMemsetAsync(agg, 0, (size_t)NN * OO * 4, stream);
    k_cvt_x <<<NN * DD / 2 / 256, 256, 0, stream>>>(x, (uint32_t*)xb);
    k_pack  <<<11 * 8192 / 256, 256, 0, stream>>>(wrel, w1W, m1W, m2W,
                                                  wrelb, w1b, m1b, m2b);
    k_project<<<dim3((ROW_TILES + 7) / 8, 16), 256, 0, stream>>>(xb, wrelb, hall);
    k_edges <<<EE * 32 / 256, 256, 0, stream>>>(src, dst, et, hall, agg);
    k_mlp   <<<(ROW_TILES + 3) / 4, 128, 0, stream>>>(xb, w1b, m1b, m2b,
                                                      w1b_, m1b_, m2b_, agg, out);
}
